// Backbone_Global_Local_17420387352869
// MI455X (gfx1250) — matrix operand order repaired and checked by reading, not compiled, not run on hardware
//
#include <hip/hip_runtime.h>
#include <hip/hip_bf16.h>
#include <cstddef>

typedef __attribute__((ext_vector_type(16))) _Float16 v16h;
typedef __attribute__((ext_vector_type(8)))  _Float16 v8h;
typedef __attribute__((ext_vector_type(8)))  float    v8f;

// CDNA5 async global->LDS path (guarded; falls back to ds_store staging)
#if defined(__has_builtin)
#  if __has_builtin(__builtin_amdgcn_global_load_async_to_lds_b128) && \
      __has_builtin(__builtin_amdgcn_s_wait_asynccnt)
#    define USE_ASYNC_LDS 1
#  endif
#endif
#ifndef USE_ASYNC_LDS
#  define USE_ASYNC_LDS 0
#endif

// ---------------------------------------------------------------------------
// Param-pointer structs (host side)
// ---------------------------------------------------------------------------
struct BNp  { const float *b, *g, *m, *v; };
struct IRp  { BNp bn1, bn2; const float* p; BNp sc_bn; const float* sc_w;
              const float *w1, *w2; bool has_sc; };
struct CropP { const float* b; IRp blk[4]; const float* w; };

// ---------------------------------------------------------------------------
// Packing / elementwise kernels
// ---------------------------------------------------------------------------
__global__ void k_pack_img(const float* __restrict__ in, _Float16* __restrict__ out,
                           int N, int C, int H, int W, int Cpad) {
  size_t i = (size_t)blockIdx.x * blockDim.x + threadIdx.x;
  size_t total = (size_t)N * H * W * Cpad;
  if (i >= total) return;
  int c = (int)(i % Cpad);
  size_t t = i / Cpad;
  int x = (int)(t % W); t /= W;
  int y = (int)(t % H);
  int n = (int)(t / H);
  float v = (c < C) ? in[(((size_t)n * C + c) * H + y) * W + x] : 0.0f;
  out[i] = (_Float16)v;
}

// OIHW fp32 -> [ky][kx][O][Ipad] fp16 (Ipad = roundup32(I), zero padded)
__global__ void k_pack_w(const float* __restrict__ w, _Float16* __restrict__ out,
                         int O, int I, int K, int Ipad) {
  size_t i = (size_t)blockIdx.x * blockDim.x + threadIdx.x;
  size_t total = (size_t)K * K * O * Ipad;
  if (i >= total) return;
  int ci = (int)(i % Ipad);
  size_t t = i / Ipad;
  int o  = (int)(t % O); t /= O;
  int kx = (int)(t % K);
  int ky = (int)(t / K);
  float v = (ci < I) ? w[(((size_t)o * I + ci) * K + ky) * K + kx] : 0.0f;
  out[i] = (_Float16)v;
}

// y = (x - m) * g/sqrt(v+eps) + b   (pre-activation BN on fp16 NHWC)
__global__ void k_bn_pre(const _Float16* __restrict__ in, _Float16* __restrict__ out,
                         size_t total, int C,
                         const float* __restrict__ g, const float* __restrict__ b,
                         const float* __restrict__ m, const float* __restrict__ v) {
  size_t i = (size_t)blockIdx.x * blockDim.x + threadIdx.x;
  if (i >= total) return;
  int c = (int)(i % C);
  float s = g[c] * rsqrtf(v[c] + 1e-5f);
  out[i] = (_Float16)(((float)in[i] - m[c]) * s + b[c]);
}

// out[n,y,x,c] = in[n, y*s, x*s, c]   (identity shortcut x[:, :, ::s, ::s])
__global__ void k_stride_copy(const _Float16* __restrict__ in, _Float16* __restrict__ out,
                              int N, int Hi, int Wi, int C, int Ho, int Wo, int s) {
  size_t i = (size_t)blockIdx.x * blockDim.x + threadIdx.x;
  size_t total = (size_t)N * Ho * Wo * C;
  if (i >= total) return;
  int c = (int)(i % C);
  size_t t = i / C;
  int x = (int)(t % Wo); t /= Wo;
  int y = (int)(t % Ho);
  int n = (int)(t / Ho);
  out[i] = in[(((size_t)n * Hi + (size_t)y * s) * Wi + (size_t)x * s) * C + c];
}

// 7x7 crop (with implicit PAD=3 zero border) of fm2 at per-image location
__global__ void k_crop_gather(const _Float16* __restrict__ fm, const int* __restrict__ locs,
                              _Float16* __restrict__ out,
                              int N, int H, int W, int C, int cropIdx) {
  size_t i = (size_t)blockIdx.x * blockDim.x + threadIdx.x;
  size_t total = (size_t)N * 7 * 7 * C;
  if (i >= total) return;
  int c = (int)(i % C);
  size_t t = i / C;
  int cx = (int)(t % 7); t /= 7;
  int cy = (int)(t % 7);
  int n  = (int)(t / 7);
  int lx = locs[((size_t)n * 5 + cropIdx) * 2 + 0];
  int ly = locs[((size_t)n * 5 + cropIdx) * 2 + 1];
  int sy = ly + cy - 3;
  int sx = lx + cx - 3;
  _Float16 v = (_Float16)0.0f;
  if (sy >= 0 && sy < H && sx >= 0 && sx < W)
    v = fm[(((size_t)n * H + sy) * W + sx) * C + c];
  out[i] = v;
}

// mean over H*W -> feature column slice of d_out (row stride FS)
__global__ void k_pool(const _Float16* __restrict__ in, float* __restrict__ outf,
                       int N, int H, int W, int C, int colBase, int FS) {
  int i = blockIdx.x * blockDim.x + threadIdx.x;
  if (i >= N * C) return;
  int n = i / C, c = i % C;
  int HW = H * W;
  float s = 0.0f;
  for (int p = 0; p < HW; ++p) s += (float)in[((size_t)n * HW + p) * C + c];
  outf[(size_t)n * FS + colBase + c] = s / (float)HW;
}

// pred = feature @ fc_w + fc_b ; loc_pred = feature[:,64:] @ loc_w + loc_b
__global__ void k_fc(const float* __restrict__ feat,
                     const float* __restrict__ fc_w, const float* __restrict__ fc_b,
                     const float* __restrict__ loc_w, const float* __restrict__ loc_b,
                     float* __restrict__ outp, int N) {
  int i = blockIdx.x * blockDim.x + threadIdx.x;
  if (i >= N * 14) return;
  int b = i / 14, j = i % 14;
  const int PRED = N * 384;
  const int LOC  = N * 384 + N * 7;
  if (j < 7) {
    float s = fc_b[j];
    for (int k2 = 0; k2 < 384; ++k2) s += feat[(size_t)b * 384 + k2] * fc_w[(size_t)k2 * 7 + j];
    outp[PRED + b * 7 + j] = s;
  } else {
    int jj = j - 7;
    float s = loc_b[jj];
    for (int k2 = 0; k2 < 320; ++k2) s += feat[(size_t)b * 384 + 64 + k2] * loc_w[(size_t)k2 * 7 + jj];
    outp[LOC + b * 7 + jj] = s;
  }
}

// ---------------------------------------------------------------------------
// Workgroup-tiled implicit-GEMM convolution on v_wmma_f32_16x16x32_f16.
//
// Block = 256 threads (8 waves) computes a 64-pixel x 64-channel output tile.
// Wave w: pixel sub-tile (w>>1)*16, channel pair (w&1)*32 -> 2 WMMAs / chunk.
// K loop over KH*KW*Ci in chunks of 32, double-buffered through LDS with the
// chunk loop manually unrolled x2 so each phase uses compile-time buffer
// indices (4 static WMMAs, no ping-pong arithmetic, no per-chunk division:
// (kk, cb) advance incrementally).
//   As[2][64 pix][32 ci], Bs[2][64 co][32 ci]  (rows padded to 40 halfs = 80 B,
//   16 B aligned for ds_load_b128, bank-spread: 20 words/row, gcd(20,64)=4).
// Staging: thread t owns (row t>>2, 8 halfs at (t&3)*8). Async global->LDS
// (ASYNCcnt) when available, else global->VGPR->ds_store_b128.
// One __syncthreads per chunk (double buffer makes the WAR barrier redundant).
//
// Fragment layout per CDNA5 ISA 7.12.2:
//   A (16x32, MxK): lanes 0-15 -> M=lane, K={0..7,16..23}; lanes 16-31 -> K={8..15,24..31}
//   B (32x16, KxN): lanes 0-15 -> N=lane, K={0..7,16..23}; lanes 16-31 -> K={8..15,24..31}
//   C/D (16x16 f32): VGPR r -> M=r (lanes 0-15) / r+8 (lanes 16-31), N=lane&15
// Fused epilogue: optional BN-affine, bias, PReLU, ReLU, residual-add.
// ---------------------------------------------------------------------------
__global__ __launch_bounds__(256) void k_conv_wmma(
    const _Float16* __restrict__ in, const _Float16* __restrict__ wp,
    _Float16* __restrict__ out,
    int N, int Hi, int Wi, int Ci,
    int Ho, int Wo, int Co,
    int KH, int KW, int stride, int pad,
    const float* __restrict__ bn_g, const float* __restrict__ bn_b,
    const float* __restrict__ bn_m, const float* __restrict__ bn_v,
    const float* __restrict__ prelu_a, const float* __restrict__ bias, int relu,
    const _Float16* __restrict__ resid) {
  __shared__ __align__(16) _Float16 As[2][64][40];
  __shared__ __align__(16) _Float16 Bs[2][64][40];

  const int tid  = threadIdx.x;
  const long long mt  = (long long)blockIdx.x * 64;   // pixel tile base
  const int ct0 = blockIdx.y * 64;                    // channel tile base
  const long long NP = (long long)N * Ho * Wo;
  const int numChunks = KH * KW * (Ci >> 5);
  const v8h z8 = {};

  // ---- staging role: thread t owns row t>>2, 8 halfs at (t&3)*8 ----
  const int sPix = tid >> 2;
  const int sOff = (tid & 3) * 8;
  const long long pg = mt + sPix;
  const bool srValid = pg < NP;
  int sn = 0, soy = 0, sox = 0;
  if (srValid) {
    sn = (int)(pg / ((long long)Ho * Wo));
    int rem = (int)(pg - (long long)sn * Ho * Wo);
    soy = rem / Wo; sox = rem - soy * Wo;
  }
  const int siy0 = soy * stride - pad;
  const int six0 = sox * stride - pad;
  const int sco  = ct0 + sPix;                        // B row channel (< Co: Co%64==0)

  // incrementally advanced next-chunk coordinates (no division in the loop)
  int kkN = 0, cbN = 0;
  auto advance = [&]() { cbN += 32; if (cbN >= Ci) { cbN = 0; ++kkN; } };

  auto stage = [&](int kk, int cb, int pp) {
    int ky = kk / KW, kx = kk - ky * KW;              // kk <= 8: cheap
    int iy = siy0 + ky, ix = six0 + kx;
    bool av = srValid && iy >= 0 && iy < Hi && ix >= 0 && ix < Wi;
    const _Float16* ga = in + (((long long)sn * Hi + iy) * Wi + ix) * (long long)Ci + cb + sOff;
    const _Float16* gb = wp + ((long long)kk * Co + sco) * (long long)Ci + cb + sOff;
    _Float16* la = &As[pp][sPix][sOff];
    _Float16* lb = &Bs[pp][sPix][sOff];
#if USE_ASYNC_LDS
    // builtin signature (from compiler diagnostic): param0 = AS(1) int4*, param1 = AS(3) int4*
    typedef int v4i __attribute__((__vector_size__(16)));
    typedef __attribute__((address_space(1))) v4i* gp4;
    typedef __attribute__((address_space(3))) v4i* lp4;
    if (av) __builtin_amdgcn_global_load_async_to_lds_b128((gp4)(void*)ga, (lp4)(void*)la, 0, 0);
    else    *(v8h*)la = z8;
    __builtin_amdgcn_global_load_async_to_lds_b128((gp4)(void*)gb, (lp4)(void*)lb, 0, 0);
#else
    v8h ra = av ? *(const v8h*)ga : z8;
    v8h rb = *(const v8h*)gb;
    *(v8h*)la = ra;
    *(v8h*)lb = rb;
#endif
  };

  // ---- compute role: wave -> 16 pixels x 32 channels ----
  const int lane  = tid & 31;
  const int wid   = tid >> 5;
  const int pixT  = wid >> 1;           // 0..3
  const int cp    = wid & 1;            // 0..1
  const int koffA  = (lane < 16) ? 0 : 8;
  const int kbaseB = (lane < 16) ? 0 : 8;
  const int arow  = pixT * 16 + (lane & 15);
  const int brow0 = cp * 32 + (lane & 15);

  v8f acc0 = {};
  v8f acc1 = {};

  auto compute = [&](int pp) {
    v8h a0 = *(const v8h*)&As[pp][arow][koffA];
    v8h a1 = *(const v8h*)&As[pp][arow][koffA + 16];
    v16h Af = __builtin_shufflevector(a0, a1, 0,1,2,3,4,5,6,7,8,9,10,11,12,13,14,15);
    v8h b00 = *(const v8h*)&Bs[pp][brow0][kbaseB];
    v8h b01 = *(const v8h*)&Bs[pp][brow0][kbaseB + 16];
    v8h b10 = *(const v8h*)&Bs[pp][brow0 + 16][kbaseB];
    v8h b11 = *(const v8h*)&Bs[pp][brow0 + 16][kbaseB + 16];
    v16h B0 = __builtin_shufflevector(b00, b01, 0,1,2,3,4,5,6,7,8,9,10,11,12,13,14,15);
    v16h B1 = __builtin_shufflevector(b10, b11, 0,1,2,3,4,5,6,7,8,9,10,11,12,13,14,15);
    acc0 = __builtin_amdgcn_wmma_f32_16x16x32_f16(false, Af, false, B0, (short)0, acc0, false, false);
    acc1 = __builtin_amdgcn_wmma_f32_16x16x32_f16(false, Af, false, B1, (short)0, acc1, false, false);
  };

  stage(kkN, cbN, 0);                   // chunk 0 -> buffer 0
  advance();
  int c = 0;
  while (true) {
    // ---- phase 0: consume buffer 0, prefetch into buffer 1 ----
#if USE_ASYNC_LDS
    __builtin_amdgcn_s_wait_asynccnt(0);
#endif
    __syncthreads();
    if (c + 1 < numChunks) { stage(kkN, cbN, 1); advance(); }
    compute(0);
    if (++c == numChunks) break;
    // ---- phase 1: consume buffer 1, prefetch into buffer 0 ----
#if USE_ASYNC_LDS
    __builtin_amdgcn_s_wait_asynccnt(0);
#endif
    __syncthreads();
    if (c + 1 < numChunks) { stage(kkN, cbN, 0); advance(); }
    compute(1);
    if (++c == numChunks) break;
  }

  // ---- epilogue: two output channels per lane ----
  const int c0 = ct0 + cp * 32 + (lane & 15);
  const int c1 = c0 + 16;
  float s0 = 1.0f, o0 = 0.0f, s1 = 1.0f, o1 = 0.0f;
  float al0 = 0.0f, al1 = 0.0f, bi0 = 0.0f, bi1 = 0.0f;
  if (bn_g) {
    float t0 = bn_g[c0] * rsqrtf(bn_v[c0] + 1e-5f);
    float t1 = bn_g[c1] * rsqrtf(bn_v[c1] + 1e-5f);
    s0 = t0; o0 = bn_b[c0] - bn_m[c0] * t0;
    s1 = t1; o1 = bn_b[c1] - bn_m[c1] * t1;
  }
  if (prelu_a) { al0 = prelu_a[c0]; al1 = prelu_a[c1]; }
  if (bias)    { bi0 = bias[c0];    bi1 = bias[c1]; }
  const int mAdd = (lane < 16) ? 0 : 8;
  for (int r = 0; r < 8; ++r) {
    long long pidx = mt + pixT * 16 + r + mAdd;
    if (pidx >= NP) continue;
    int nn = (int)(pidx / ((long long)Ho * Wo));
    int rr = (int)(pidx - (long long)nn * Ho * Wo);
    int yy = rr / Wo, xx = rr - yy * Wo;
    long long base = (((long long)nn * Ho + yy) * Wo + xx) * (long long)Co;
    float v0 = acc0[r], v1 = acc1[r];
    if (bn_g)    { v0 = v0 * s0 + o0;               v1 = v1 * s1 + o1; }
    if (bias)    { v0 += bi0;                        v1 += bi1; }
    if (prelu_a) { v0 = (v0 >= 0.f) ? v0 : al0 * v0; v1 = (v1 >= 0.f) ? v1 : al1 * v1; }
    if (relu)    { v0 = fmaxf(v0, 0.f);              v1 = fmaxf(v1, 0.f); }
    if (resid)   { v0 += (float)resid[base + c0];    v1 += (float)resid[base + c1]; }
    out[base + c0] = (_Float16)v0;
    out[base + c1] = (_Float16)v1;
  }
}

// ---------------------------------------------------------------------------
// Host orchestration
// ---------------------------------------------------------------------------
extern "C" void kernel_launch(void* const* d_in, const int* in_sizes, int n_in,
                              void* d_out, int out_size, void* d_ws, size_t ws_size,
                              hipStream_t stream) {
  (void)in_sizes; (void)n_in; (void)out_size; (void)ws_size;
  const int N = 64;
  const float* imgs = (const float*)d_in[0];
  const int*   locs = (const int*)d_in[1];

  // ---- walk param leaves in JAX pytree order (dict keys sorted) ----
  int li = 2;
  auto F = [&]() -> const float* { return (const float*)d_in[li++]; };
  auto readBN = [&]() { BNp x; x.b = F(); x.g = F(); x.m = F(); x.v = F(); return x; };
  auto readIR = [&](bool sc) {
    IRp u{}; u.has_sc = sc;
    u.bn1 = readBN(); u.bn2 = readBN(); u.p = F();
    if (sc) { u.sc_bn = readBN(); u.sc_w = F(); }
    u.w1 = F(); u.w2 = F();
    return u;
  };

  CropP crop[5];
  for (int i = 0; i < 5; ++i) {
    crop[i].b = F();
    for (int j = 0; j < 4; ++j) crop[i].blk[j] = readIR(j == 0 || j == 2);
    crop[i].w = F();
  }
  const float* fc_b = F(); const float* fc_w = F();
  BNp in_bn = readBN(); const float* in_p = F(); const float* in_w = F();

  const int bn_units[4] = {3, 4, 14, 3};
  IRp trunk[24];
  int tCi[24], tCo[24], tS[24];
  {
    const int bci[4] = {64, 64, 128, 256}, bd[4] = {64, 128, 256, 512};
    int k = 0;
    for (int b = 0; b < 4; ++b)
      for (int u = 0; u < bn_units[b]; ++u) {
        trunk[k] = readIR(u == 0 && b != 0);   // block0 ci==d even at stride 2
        tCi[k] = (u == 0) ? bci[b] : bd[b];
        tCo[k] = bd[b];
        tS[k]  = (u == 0) ? 2 : 1;
        ++k;
      }
  }
  const float* loc_b = F(); const float* loc_w = F();
  const float* out_b = F(); const float* out_w = F();

  // ---- workspace arena ----
  char* wsp = (char*)d_ws; size_t woff = 0;
  auto walloc = [&](size_t bytes) -> void* {
    void* p = wsp + woff; woff += (bytes + 255) & ~(size_t)255; return p;
  };
  _Float16* wpack = (_Float16*)walloc((size_t)512 * 512 * 9 * 2);   // max packed conv
  _Float16* ximg  = (_Float16*)walloc((size_t)N * 112 * 112 * 32 * 2);
  _Float16* fm2   = (_Float16*)walloc((size_t)N * 28 * 28 * 128 * 2);
  _Float16* ring[4];
  for (int i = 0; i < 4; ++i) ring[i] = (_Float16*)walloc((size_t)N * 112 * 112 * 64 * 2);

  // ---- conv helper: on-device weight pack + tiled WMMA implicit GEMM ----
  auto conv = [&](const _Float16* in, const float* wraw, _Float16* out,
                  int Hi, int Wi, int Ci, int Co, int K, int s, int pad,
                  const BNp* bn, const float* pa, const float* bias, int relu,
                  const _Float16* resid, int* HoOut, int* WoOut) {
    int Ipad = (Ci + 31) & ~31;
    size_t wt = (size_t)K * K * Co * Ipad;
    k_pack_w<<<dim3((unsigned)((wt + 255) / 256)), 256, 0, stream>>>(wraw, wpack, Co, Ci, K, Ipad);
    int Ho = (Hi + 2 * pad - K) / s + 1;
    int Wo = (Wi + 2 * pad - K) / s + 1;
    size_t NP = (size_t)N * Ho * Wo;
    dim3 grid((unsigned)((NP + 63) / 64), (unsigned)(Co / 64));
    k_conv_wmma<<<grid, 256, 0, stream>>>(
        in, wpack, out, N, Hi, Wi, Ipad, Ho, Wo, Co, K, K, s, pad,
        bn ? bn->g : nullptr, bn ? bn->b : nullptr, bn ? bn->m : nullptr, bn ? bn->v : nullptr,
        pa, bias, relu, resid);
    if (HoOut) *HoOut = Ho;
    if (WoOut) *WoOut = Wo;
  };

  // ---- one IR unit (pre-act residual, 2x 3x3 convs) ----
  auto run_ir = [&](const IRp& u, int& xi, int& H, int& W, int Ci, int Co, int s) {
    int oth[3], t = 0;
    for (int i = 0; i < 4; ++i) if (i != xi) oth[t++] = i;
    _Float16* A  = ring[oth[0]];
    _Float16* Bf = ring[oth[1]];
    _Float16* Cc = ring[oth[2]];
    _Float16* x  = ring[xi];

    size_t tin = (size_t)N * H * W * Ci;
    k_bn_pre<<<(unsigned)((tin + 255) / 256), 256, 0, stream>>>(
        x, A, tin, Ci, u.bn1.g, u.bn1.b, u.bn1.m, u.bn1.v);
    // r2 = prelu(conv3x3_s1(r, w1))
    conv(A, u.w1, Bf, H, W, Ci, Co, 3, 1, 1, nullptr, u.p, nullptr, 0, nullptr, nullptr, nullptr);
    // shortcut
    const _Float16* resid;
    if (u.has_sc) {
      conv(x, u.sc_w, Cc, H, W, Ci, Co, 1, s, 0, &u.sc_bn, nullptr, nullptr, 0, nullptr, nullptr, nullptr);
      resid = Cc;
    } else if (s != 1) {
      int Ho = (H - 1) / s + 1, Wo = (W - 1) / s + 1;
      size_t to = (size_t)N * Ho * Wo * Ci;
      k_stride_copy<<<(unsigned)((to + 255) / 256), 256, 0, stream>>>(x, Cc, N, H, W, Ci, Ho, Wo, s);
      resid = Cc;
    } else {
      resid = x;
    }
    // out = bn2(conv3x3_s(r2, w2)) + sc
    int Ho, Wo;
    conv(Bf, u.w2, A, H, W, Co, Co, 3, s, 1, &u.bn2, nullptr, nullptr, 0, resid, &Ho, &Wo);
    xi = oth[0]; H = Ho; W = Wo;
  };

  // ================= pipeline =================
  {
    size_t ti = (size_t)N * 112 * 112 * 32;
    k_pack_img<<<(unsigned)((ti + 255) / 256), 256, 0, stream>>>(imgs, ximg, N, 3, 112, 112, 32);
  }
  int xi = 0, H = 112, W = 112;
  conv(ximg, in_w, ring[0], 112, 112, 3, 64, 3, 1, 1, &in_bn, in_p, nullptr, 0, nullptr, nullptr, nullptr);

  int ui = 0;
  for (int b = 0; b < 4; ++b) {
    for (int u = 0; u < bn_units[b]; ++u) { run_ir(trunk[ui], xi, H, W, tCi[ui], tCo[ui], tS[ui]); ++ui; }
    if (b == 1)  // snapshot fm2 = [64,28,28,128] for the crop branches
      (void)hipMemcpyAsync(fm2, ring[xi], (size_t)N * 28 * 28 * 128 * 2,
                           hipMemcpyDeviceToDevice, stream);
  }

  // global head: relu(conv3x3(x512@7x7) + out_b) -> mean -> feature[:, 0:64]
  float* outF = (float*)d_out;
  {
    int o0 = -1; for (int i = 0; i < 4; ++i) if (i != xi) { o0 = i; break; }
    conv(ring[xi], out_w, ring[o0], 7, 7, 512, 64, 3, 1, 1,
         nullptr, nullptr, out_b, 1, nullptr, nullptr, nullptr);
    k_pool<<<(unsigned)((N * 64 + 255) / 256), 256, 0, stream>>>(ring[o0], outF, N, 7, 7, 64, 0, 384);
  }

  // crop branches
  static const int cc[4][3] = {{128, 256, 2}, {256, 256, 1}, {256, 512, 2}, {512, 512, 1}};
  for (int i = 0; i < 5; ++i) {
    int cxi = 0, Hc = 7, Wc = 7;
    size_t tc = (size_t)N * 7 * 7 * 128;
    k_crop_gather<<<(unsigned)((tc + 255) / 256), 256, 0, stream>>>(fm2, locs, ring[0], N, 28, 28, 128, i);
    for (int j = 0; j < 4; ++j) run_ir(crop[i].blk[j], cxi, Hc, Wc, cc[j][0], cc[j][1], cc[j][2]);
    int o0 = -1; for (int q = 0; q < 4; ++q) if (q != cxi) { o0 = q; break; }
    conv(ring[cxi], crop[i].w, ring[o0], Hc, Wc, 512, 64, 3, 1, 1,
         nullptr, nullptr, crop[i].b, 1, nullptr, nullptr, nullptr);
    k_pool<<<(unsigned)((N * 64 + 255) / 256), 256, 0, stream>>>(ring[o0], outF, N, Hc, Wc, 64, 64 + 64 * i, 384);
  }

  // FC heads (read feature region of d_out, write pred / loc_pred regions)
  k_fc<<<(unsigned)((N * 14 + 255) / 256), 256, 0, stream>>>(outF, fc_w, fc_b, loc_w, loc_b, outF, N);
}
